// RandomProjectionQuantizer_24704651886985
// MI455X (gfx1250) — matrix operand order repaired and checked by reading, not compiled, not run on hardware
//
#include <hip/hip_runtime.h>

// ---------------------------------------------------------------------------
// RandomProjectionQuantizer for MI455X (gfx1250, wave32, WMMA bf16 16x16x32)
//   proj = x @ W ; nx = rownorm(proj) ; nearest = argmax_k (ncb @ nx^T)
// ---------------------------------------------------------------------------

typedef __attribute__((ext_vector_type(16))) __bf16 v16bf;
typedef __attribute__((ext_vector_type(8)))  __bf16 v8bf;
typedef __attribute__((ext_vector_type(8)))  float  v8f;

#define D_IN       1024      // input dim (K of projection GEMM)
#define E_DIM      128       // codebook dim
#define K_CB       8192      // codebook size
#define M_TILE     128       // rows per workgroup (8 waves x 16 rows)
#define NXS_STRIDE 136       // 128 + 8 bf16 pad -> conflict-free b128 LDS loads

// native RNE f32 -> bf16 (lowers to v_cvt_pk_bf16_f32)
static __device__ __forceinline__ __bf16 f2bf(float f) {
  return (__bf16)f;
}

static __device__ __forceinline__ v16bf cvt16(v8f f0, v8f f1) {
  v8bf lo = __builtin_convertvector(f0, v8bf);
  v8bf hi = __builtin_convertvector(f1, v8bf);
  return __builtin_shufflevector(lo, hi, 0,1,2,3,4,5,6,7,8,9,10,11,12,13,14,15);
}

static __device__ __forceinline__ v8f wmma_bf16(v16bf a, v16bf b, v8f c) {
  // D = A(16x32 bf16) x B(32x16 bf16) + C(16x16 f32)
  return __builtin_amdgcn_wmma_f32_16x16x32_bf16(
      /*neg_a=*/false, a, /*neg_b=*/false, b,
      /*c_mod=*/(short)0, c, /*reuse_a=*/false, /*reuse_b=*/false);
}

// --------------------------- prep: W -> bf16 -------------------------------
__global__ void rpq_prep_w(const float* __restrict__ w, __bf16* __restrict__ wbf) {
  int i = blockIdx.x * blockDim.x + threadIdx.x;   // exactly 1024*128 threads
  wbf[i] = f2bf(w[i]);
}

// ------------- prep: normalize codebook rows, store transposed -------------
// ncbT[k][n] = codebook[n][k] / max(||codebook[n]||, 1e-12), bf16, [128][8192]
__global__ void rpq_prep_cb(const float* __restrict__ cb, __bf16* __restrict__ ncbT) {
  int wv   = threadIdx.x >> 5;
  int lane = threadIdx.x & 31;
  int r    = blockIdx.x * 8 + wv;                  // 1024 blocks * 8 waves = 8192 rows
  const float* row = cb + (size_t)r * E_DIM;
  float4 f = *(const float4*)(row + lane * 4);
  float s = f.x * f.x + f.y * f.y + f.z * f.z + f.w * f.w;
  s += __shfl_xor(s, 1);
  s += __shfl_xor(s, 2);
  s += __shfl_xor(s, 4);
  s += __shfl_xor(s, 8);
  s += __shfl_xor(s, 16);
  float sc = 1.0f / fmaxf(sqrtf(s), 1e-12f);
  int k = lane * 4;
  ncbT[(size_t)(k + 0) * K_CB + r] = f2bf(f.x * sc);
  ncbT[(size_t)(k + 1) * K_CB + r] = f2bf(f.y * sc);
  ncbT[(size_t)(k + 2) * K_CB + r] = f2bf(f.z * sc);
  ncbT[(size_t)(k + 3) * K_CB + r] = f2bf(f.w * sc);
}

// ------------------------------- main --------------------------------------
__global__ __launch_bounds__(256)
void rpq_main(const float* __restrict__ x, const __bf16* __restrict__ wbf,
              const __bf16* __restrict__ ncbT, int* __restrict__ out) {
  __shared__ __align__(16) __bf16 nxs[M_TILE * NXS_STRIDE];   // ~34 KB

  const int tid  = threadIdx.x;
  const int wv   = tid >> 5;        // wave id (0..7) -> owns rows 16*wv..16*wv+15
  const int lane = tid & 31;
  const int m    = lane & 15;       // row (A) / col (B,C,D) index within tile
  const int hlf  = lane >> 4;       // lane half
  const int tile = blockIdx.x;      // 256 tiles of 128 rows

  // ===== Phase 1: P[16x128] = X_tile[16x1024] @ W[1024x128]  (per wave) =====
  v8f acc[8];
  #pragma unroll
  for (int t = 0; t < 8; ++t) acc[t] = (v8f){0.f,0.f,0.f,0.f,0.f,0.f,0.f,0.f};

  const int rowg = tile * M_TILE + wv * 16 + m;
  const float* xrow = x + (size_t)rowg * D_IN;

  // x2 unroll -> alternating A-fragment registers, kills WMMA->cvt WAR nops
  #pragma unroll 2
  for (int kk = 0; kk < D_IN; kk += 32) {
    // A fragment (16x32 bf16): lane m holds row m; half 0 -> K {0..7,16..23},
    // half 1 -> K {8..15,24..31}; two 32B f32 loads, packed cvt to bf16.
    v8f f0 = *(const v8f*)(xrow + kk + hlf * 8);
    v8f f1 = *(const v8f*)(xrow + kk + 16 + hlf * 8);
    v16bf a = cvt16(f0, f1);

    // B fragment: lane holds W column N = t*16+m, K in same order as A
    const __bf16* bp = wbf + (size_t)kk * E_DIM + m;
    #pragma unroll
    for (int t = 0; t < 8; ++t) {
      v16bf b;
      #pragma unroll
      for (int s = 0; s < 16; ++s)
        b[s] = bp[(size_t)((s < 8 ? 8 * hlf + s : 16 + 8 * hlf + s - 8)) * E_DIM + t * 16];
      acc[t] = wmma_bf16(a, b, acc[t]);
    }
  }

  // ===== Normalize rows (C layout: lane holds col m, rows v + 8*hlf) =====
  #pragma unroll
  for (int v = 0; v < 8; ++v) {
    float s = 0.f;
    #pragma unroll
    for (int t = 0; t < 8; ++t) s += acc[t][v] * acc[t][v];
    s += __shfl_xor(s, 1);
    s += __shfl_xor(s, 2);
    s += __shfl_xor(s, 4);
    s += __shfl_xor(s, 8);                         // 16-lane half-butterfly
    float sc = 1.0f / fmaxf(sqrtf(s), 1e-12f);
    int lrow = wv * 16 + v + 8 * hlf;
    __bf16* dst = &nxs[lrow * NXS_STRIDE + m];
    #pragma unroll
    for (int t = 0; t < 8; ++t) dst[t * 16] = f2bf(acc[t][v] * sc);
  }
  __syncthreads();

  // ===== Phase 2: scores = NX[16x128] @ ncbT[128x8192], running argmax =====
  v16bf afr[4];                                     // A resident: 16 rows x K=128
  const __bf16* nrow = &nxs[(wv * 16 + m) * NXS_STRIDE];
  #pragma unroll
  for (int ks = 0; ks < 4; ++ks) {
    v8bf lo = *(const v8bf*)(nrow + ks * 32 + hlf * 8);
    v8bf hi = *(const v8bf*)(nrow + ks * 32 + 16 + hlf * 8);
    afr[ks] = __builtin_shufflevector(lo, hi,
              0,1,2,3,4,5,6,7,8,9,10,11,12,13,14,15);
  }

  float best[8]; int besti[8];
  #pragma unroll
  for (int v = 0; v < 8; ++v) { best[v] = -3.4e38f; besti[v] = 0; }

  // x2 unroll -> two independent WMMA accumulator chains + load streams
  #pragma unroll 2
  for (int n0 = 0; n0 < K_CB; n0 += 16) {           // 512 codebook tiles
    v8f d = (v8f){0.f,0.f,0.f,0.f,0.f,0.f,0.f,0.f};
    #pragma unroll
    for (int ks = 0; ks < 4; ++ks) {
      v16bf b;
      #pragma unroll
      for (int s = 0; s < 16; ++s)
        b[s] = ncbT[(size_t)(ks * 32 + (s < 8 ? 8 * hlf + s : 16 + 8 * hlf + s - 8)) * K_CB + n0 + m];
      d = wmma_bf16(afr[ks], b, d);
    }
    const int col = n0 + m;
    #pragma unroll
    for (int v = 0; v < 8; ++v) {                   // strict > keeps lowest index
      if (d[v] > best[v]) { best[v] = d[v]; besti[v] = col; }
    }
  }

  // ===== Final 16-lane argmax reduce (ties -> lower index), write int32 =====
  #pragma unroll
  for (int v = 0; v < 8; ++v) {
    float s = best[v]; int i = besti[v];
    #pragma unroll
    for (int msk = 1; msk <= 8; msk <<= 1) {
      float os = __shfl_xor(s, msk);
      int   oi = __shfl_xor(i, msk);
      if (os > s || (os == s && oi < i)) { s = os; i = oi; }
    }
    if (m == 0) out[tile * M_TILE + wv * 16 + v + 8 * hlf] = i;
  }
}

// ------------------------------ launcher -----------------------------------
extern "C" void kernel_launch(void* const* d_in, const int* in_sizes, int n_in,
                              void* d_out, int out_size, void* d_ws, size_t ws_size,
                              hipStream_t stream) {
  const float* x  = (const float*)d_in[0];   // [8,4096,1024]
  const float* w  = (const float*)d_in[1];   // [1024,128]
  const float* cb = (const float*)d_in[2];   // [8192,128]
  int* out = (int*)d_out;                    // [8,4096] int32

  __bf16* wbf  = (__bf16*)d_ws;                                     // 256 KB
  __bf16* ncbT = (__bf16*)((char*)d_ws + (size_t)D_IN * E_DIM * 2); // 2 MB

  rpq_prep_w <<<(D_IN * E_DIM) / 256, 256, 0, stream>>>(w, wbf);
  rpq_prep_cb<<<K_CB / 8,            256, 0, stream>>>(cb, ncbT);
  rpq_main   <<<(8 * 4096) / M_TILE, 256, 0, stream>>>(x, wbf, ncbT, out);
}
